// MultiAlternatingTransformer_4870492914034
// MI455X (gfx1250) — matrix operand order repaired and checked by reading, not compiled, not run on hardware
//
#include <hip/hip_runtime.h>
#include <hip/hip_bf16.h>

typedef __attribute__((ext_vector_type(16))) __bf16 v16bf;
typedef __attribute__((ext_vector_type(8)))  __bf16 v8bf;
typedef __attribute__((ext_vector_type(8)))  float  v8f;

#define WMMA_BF16(a, b, c) \
  __builtin_amdgcn_wmma_f32_16x16x32_bf16(false, (a), false, (b), (short)0, (c), false, false)

// async 16-byte global -> LDS copy (CDNA5 ASYNCcnt path)
#define ASYNC_CP16(ldsPtr, gPtr)                                               \
  asm volatile("global_load_async_to_lds_b128 %0, %1, off"                     \
               :: "v"((unsigned)(unsigned long long)(ldsPtr)), "v"(gPtr)       \
               : "memory")
#define WAIT_ASYNC0() asm volatile("s_wait_asynccnt 0" ::: "memory")
#define WAIT_ASYNC3() asm volatile("s_wait_asynccnt 3" ::: "memory")
#define WAIT_DS()     asm volatile("s_wait_dscnt 0" ::: "memory")

#define NN    4096
#define DD    256
#define HH    8
#define CC    32
#define EE    131072
#define EDIMM 16
#define GG    64
#define LL    64

// build v16bf from 16 contiguous LDS bf16 (2x ds_load_b128)
__device__ inline v16bf load16(const __bf16* p) {
  v8bf lo = *(const v8bf*)p;
  v8bf hi = *(const v8bf*)(p + 8);
  return __builtin_shufflevector(lo, hi, 0,1,2,3,4,5,6,7,8,9,10,11,12,13,14,15);
}
// build v16bf from 8 at p and 8 at p+16 (A-fragment pattern)
__device__ inline v16bf load8x2(const __bf16* p) {
  v8bf lo = *(const v8bf*)p;
  v8bf hi = *(const v8bf*)(p + 16);
  return __builtin_shufflevector(lo, hi, 0,1,2,3,4,5,6,7,8,9,10,11,12,13,14,15);
}

// ---------------------------------------------------------------------------
// utility kernels
// ---------------------------------------------------------------------------
__global__ void mat_f32_to_bf16(const float* __restrict__ src, __bf16* __restrict__ dst, int n) {
  int i = blockIdx.x * 256 + threadIdx.x;
  if (i < n) dst[i] = (__bf16)src[i];
}

// W [K x 256] f32 -> Wt [256 x K] bf16 (transposed for contiguous B-frags)
__global__ void wt_transpose_bf16(const float* __restrict__ src, __bf16* __restrict__ dst,
                                  int Kdim) {
  int i = blockIdx.x * 256 + threadIdx.x;
  if (i >= Kdim * DD) return;
  int k = i >> 8, c = i & 255;
  dst[(size_t)c * Kdim + k] = (__bf16)src[i];
}

// V [N x 256] f32 -> Vt [256 x N] bf16
__global__ void v_transpose_bf16(const float* __restrict__ src, __bf16* __restrict__ dst) {
  int i = blockIdx.x * 256 + threadIdx.x;
  int nnode = i >> 8, d = i & 255;
  dst[(size_t)d * NN + nnode] = (__bf16)src[i];
}

__global__ void fill_f32(float* __restrict__ p, float v, int n) {
  int i = blockIdx.x * 256 + threadIdx.x;
  if (i < n) p[i] = v;
}

// ---------------------------------------------------------------------------
// Y[N x 256] = Xb[N x K] @ W[K x 256] + bias, W given transposed (Wt: 256 x K)
// 256 thr = 8 waves; block tile 128x64, wave tile 32x32 (4 WMMA / K-step).
// Double-buffered LDS + software-pipelined async global->LDS staging.
// ---------------------------------------------------------------------------
__global__ __launch_bounds__(256)
void gemm_bias_wmma(const __bf16* __restrict__ X, const __bf16* __restrict__ Wt,
                    const float* __restrict__ bias, float* __restrict__ Y,
                    int Kdim) {
  __shared__ alignas(16) __bf16 sX[2][128 * 32];    // [row][k]
  __shared__ alignas(16) __bf16 sWt[2][64 * 32];    // [col][k]
  const int tid  = threadIdx.x;
  const int wave = tid >> 5, lane = tid & 31;
  const int rowBase = blockIdx.x * 128;
  const int colBase = blockIdx.y * 64;
  const int wr = (wave >> 1) * 32;
  const int wc = (wave & 1) * 32;
  const int m = lane & 15, half = lane >> 4;

  // 3 async 16B chunks per thread per stage (X: 512 chunks, W: 256 chunks)
  auto stage = [&](int buf, int k0) {
    for (int t = tid; t < 512; t += 256) {
      int r = t >> 2, j = t & 3;
      ASYNC_CP16(&sX[buf][r * 32 + j * 8], X + (size_t)(rowBase + r) * Kdim + k0 + j * 8);
    }
    {
      int r = tid >> 2, j = tid & 3;
      ASYNC_CP16(&sWt[buf][r * 32 + j * 8], Wt + (size_t)(colBase + r) * Kdim + k0 + j * 8);
    }
  };

  v8f acc00 = {}, acc01 = {}, acc10 = {}, acc11 = {};
  const int nk = Kdim >> 5;
  stage(0, 0);
  for (int ks = 0; ks < nk; ++ks) {
    const int buf = ks & 1;
    if (ks + 1 < nk) {
      stage(buf ^ 1, (ks + 1) << 5);
      WAIT_ASYNC3();                 // current tile resident; next 3 still in flight
    } else {
      WAIT_ASYNC0();
    }
    __syncthreads();

    v16bf a0 = load8x2(&sX[buf][(wr + m) * 32 + half * 8]);
    v16bf a1 = load8x2(&sX[buf][(wr + 16 + m) * 32 + half * 8]);
    v16bf b0 = load8x2(&sWt[buf][(wc + m) * 32 + half * 8]);
    v16bf b1 = load8x2(&sWt[buf][(wc + 16 + m) * 32 + half * 8]);
    acc00 = WMMA_BF16(a0, b0, acc00);
    acc01 = WMMA_BF16(a0, b1, acc01);
    acc10 = WMMA_BF16(a1, b0, acc10);
    acc11 = WMMA_BF16(a1, b1, acc11);
    __syncthreads();
  }
#pragma unroll
  for (int rr = 0; rr < 8; ++rr) {
    int row0 = rowBase + wr + half * 8 + rr;
    int row1 = row0 + 16;
    int c0   = colBase + wc + m;
    int c1   = c0 + 16;
    float b0v = bias[c0], b1v = bias[c1];
    Y[(size_t)row0 * DD + c0] = acc00[rr] + b0v;
    Y[(size_t)row0 * DD + c1] = acc01[rr] + b1v;
    Y[(size_t)row1 * DD + c0] = acc10[rr] + b0v;
    Y[(size_t)row1 * DD + c1] = acc11[rr] + b1v;
  }
}

// ---------------------------------------------------------------------------
// dense block attention (layers 1,3): one wave per (graph, head)
// S = Q K^T / sqrt(C); softmax over src; O = P V  (V given transposed D x N)
// ---------------------------------------------------------------------------
__global__ __launch_bounds__(32)
void dense_attn_wmma(const __bf16* __restrict__ qb, const __bf16* __restrict__ kb,
                     const __bf16* __restrict__ vbt, float* __restrict__ out) {
  __shared__ alignas(16) __bf16 sQ[64 * 32];    // [src][c]
  __shared__ alignas(16) __bf16 sK[64 * 32];    // [src][c]
  __shared__ alignas(16) __bf16 sVt[32 * 64];   // [c][src]
  __shared__ alignas(16) __bf16 sP[16 * 64];
  const int g = blockIdx.x, h = blockIdx.y;
  const int base = g * LL;
  const int lane = threadIdx.x;
  const int m = lane & 15, half = lane >> 4;
  const float scale = 0.17677669529663687f;  // 1/sqrt(32)

  for (int t = lane; t < 256; t += 32) {       // 16B chunks
    int rr = t >> 2, jj = t & 3;
    size_t off = (size_t)(base + rr) * DD + h * CC + jj * 8;
    ASYNC_CP16(&sQ[rr * 32 + jj * 8], qb + off);
    ASYNC_CP16(&sK[rr * 32 + jj * 8], kb + off);
    int c = t >> 3, js = t & 7;
    ASYNC_CP16(&sVt[c * 64 + js * 8], vbt + (size_t)(h * CC + c) * NN + base + js * 8);
  }
  WAIT_ASYNC0();

  for (int mt = 0; mt < 4; ++mt) {
    v16bf aq = load8x2(&sQ[(mt * 16 + m) * 32 + half * 8]);
    v8f s[4];
#pragma unroll
    for (int nt = 0; nt < 4; ++nt) {
      v16bf bk = load8x2(&sK[(nt * 16 + m) * 32 + half * 8]);  // B[kk][n]=K[src][kk]
      v8f z = {};
      s[nt] = WMMA_BF16(aq, bk, z);
    }
    // per-row softmax across src (16 lanes in half + 4 accumulators)
    float rowsum[8];
#pragma unroll
    for (int rr = 0; rr < 8; ++rr) {
      float mx = -3.0e38f;
#pragma unroll
      for (int nt = 0; nt < 4; ++nt) mx = fmaxf(mx, s[nt][rr] * scale);
      for (int sh = 1; sh < 16; sh <<= 1) mx = fmaxf(mx, __shfl_xor(mx, sh, 32));
      float sum = 0.f;
#pragma unroll
      for (int nt = 0; nt < 4; ++nt) {
        float p = __expf(s[nt][rr] * scale - mx);
        s[nt][rr] = p;
        sum += p;
      }
      for (int sh = 1; sh < 16; sh <<= 1) sum += __shfl_xor(sum, sh, 32);
      rowsum[rr] = sum;
    }
#pragma unroll
    for (int rr = 0; rr < 8; ++rr) {
      float inv = 1.0f / rowsum[rr];
      int M = half * 8 + rr;
#pragma unroll
      for (int nt = 0; nt < 4; ++nt)
        sP[M * 64 + nt * 16 + m] = (__bf16)(s[nt][rr] * inv);
    }
    WAIT_DS();
    // O(16x32) = P(16x64) @ V(64x32)
    v8f o0 = {}, o1 = {};
#pragma unroll
    for (int kc = 0; kc < 2; ++kc) {
      v16bf ap  = load8x2(&sP[m * 64 + kc * 32 + half * 8]);
      v16bf bv0 = load8x2(&sVt[m * 64 + kc * 32 + half * 8]);
      v16bf bv1 = load8x2(&sVt[(16 + m) * 64 + kc * 32 + half * 8]);
      o0 = WMMA_BF16(ap, bv0, o0);
      o1 = WMMA_BF16(ap, bv1, o1);
    }
#pragma unroll
    for (int rr = 0; rr < 8; ++rr) {
      int row = base + mt * 16 + half * 8 + rr;
      out[(size_t)row * DD + h * CC + m]      = o0[rr];
      out[(size_t)row * DD + h * CC + 16 + m] = o1[rr];
    }
    WAIT_DS();
  }
}

// ---------------------------------------------------------------------------
// sparse attention (layers 0,2): edge-conditioned, segment softmax via atomics
// ---------------------------------------------------------------------------
__device__ inline void atomicMaxF(float* addr, float val) {
  int* ai = (int*)addr;
  int old = *ai;
  while (val > __int_as_float(old)) {
    int assumed = old;
    old = atomicCAS(ai, assumed, __float_as_int(val));
    if (old == assumed) break;
  }
}

__global__ void sparse_alpha(const float* __restrict__ q, const float* __restrict__ k,
                             const int* __restrict__ ei, const float* __restrict__ ea,
                             const float* __restrict__ We, float* __restrict__ alpha,
                             float* __restrict__ amax) {
  int i = blockIdx.x * 256 + threadIdx.x;
  if (i >= EE * HH) return;
  int e = i >> 3, h = i & 7;
  int src = ei[e], dst = ei[EE + e];
  const float* qrow = q + (size_t)dst * DD + h * CC;
  const float* krow = k + (size_t)src * DD + h * CC;
  float av[EDIMM];
#pragma unroll
  for (int j = 0; j < EDIMM; ++j) av[j] = ea[(size_t)e * EDIMM + j];
  float acc = 0.f;
  for (int c = 0; c < CC; ++c) {
    float ec = 0.f;
#pragma unroll
    for (int j = 0; j < EDIMM; ++j) ec += av[j] * We[j * DD + h * CC + c];
    acc += qrow[c] * (krow[c] + ec);
  }
  float a = acc * 0.17677669529663687f;
  alpha[i] = a;
  atomicMaxF(&amax[dst * HH + h], a);
}

__global__ void sparse_softmax_p(const int* __restrict__ ei, float* __restrict__ alpha,
                                 const float* __restrict__ amax, float* __restrict__ denom) {
  int i = blockIdx.x * 256 + threadIdx.x;
  if (i >= EE * HH) return;
  int e = i >> 3, h = i & 7;
  int dst = ei[EE + e];
  float p = __expf(alpha[i] - amax[dst * HH + h]);
  alpha[i] = p;
  atomicAdd(&denom[dst * HH + h], p);
}

__global__ void sparse_accum(const int* __restrict__ ei, const float* __restrict__ ea,
                             const float* __restrict__ We, const float* __restrict__ v,
                             const float* __restrict__ alpha, float* __restrict__ out) {
  int i = blockIdx.x * 256 + threadIdx.x;   // EE*256 threads
  int e = i >> 8, d = i & 255;
  int h = d >> 5;
  int src = ei[e], dst = ei[EE + e];
  float p = alpha[(size_t)e * HH + h];
  float val = v[(size_t)src * DD + d];
  float ec = 0.f;
#pragma unroll
  for (int j = 0; j < EDIMM; ++j) ec += ea[(size_t)e * EDIMM + j] * We[j * DD + d];
  atomicAdd(&out[(size_t)dst * DD + d], p * (val + ec));
}

// ---------------------------------------------------------------------------
// per-node epilogue: beta gate + layernorm + leaky-relu. one block per node.
// wave32 shfl reductions + 8-entry LDS combine.
// ---------------------------------------------------------------------------
__device__ inline float block_sum256(float v, float* part, int lane, int wave) {
#pragma unroll
  for (int sh = 16; sh > 0; sh >>= 1) v += __shfl_xor(v, sh, 32);
  if (lane == 0) part[wave] = v;
  __syncthreads();
  float tot = 0.f;
#pragma unroll
  for (int w = 0; w < 8; ++w) tot += part[w];
  __syncthreads();
  return tot;
}

__global__ __launch_bounds__(256)
void node_epilogue(const float* __restrict__ attn, const float* __restrict__ denom,
                   const float* __restrict__ rbuf, const float* __restrict__ wb,
                   const float* __restrict__ lng, const float* __restrict__ lnb,
                   float* __restrict__ out) {
  __shared__ float part[8];
  const int n = blockIdx.x, d = threadIdx.x, h = d >> 5;
  const int lane = d & 31, wave = d >> 5;
  float o = attn[(size_t)n * DD + d];
  if (denom) o /= denom[n * HH + h];
  float r = rbuf[(size_t)n * DD + d];
  float logit = block_sum256(wb[d] * o + wb[DD + d] * r + wb[2 * DD + d] * (o - r),
                             part, lane, wave);
  float g = 1.f / (1.f + __expf(-logit));
  float hv = g * r + (1.f - g) * o;
  float mean = block_sum256(hv, part, lane, wave) * (1.f / 256.f);
  float cen = hv - mean;
  float var = block_sum256(cen * cen, part, lane, wave) * (1.f / 256.f);
  float y = cen * rsqrtf(var + 1e-5f) * lng[d] + lnb[d];
  out[(size_t)n * DD + d] = (y > 0.f) ? y : 0.01f * y;
}

// ---------------------------------------------------------------------------
// host orchestration
// ---------------------------------------------------------------------------
extern "C" void kernel_launch(void* const* d_in, const int* in_sizes, int n_in,
                              void* d_out, int out_size, void* d_ws, size_t ws_size,
                              hipStream_t stream) {
  const float* x      = (const float*)d_in[0];
  const float* eattr  = (const float*)d_in[1];
  const float* wq0    = (const float*)d_in[2];
  const float* bq0    = (const float*)d_in[3];
  const float* wk0    = (const float*)d_in[4];
  const float* bk0    = (const float*)d_in[5];
  const float* wv0    = (const float*)d_in[6];
  const float* bv0    = (const float*)d_in[7];
  const float* we0    = (const float*)d_in[8];
  const float* wr0    = (const float*)d_in[9];
  const float* br0    = (const float*)d_in[10];
  const float* wbeta0 = (const float*)d_in[11];
  const float* lng0   = (const float*)d_in[12];
  const float* lnb0   = (const float*)d_in[13];
  const float* wq     = (const float*)d_in[14];
  const float* bq     = (const float*)d_in[15];
  const float* wk     = (const float*)d_in[16];
  const float* bk     = (const float*)d_in[17];
  const float* wv     = (const float*)d_in[18];
  const float* bv     = (const float*)d_in[19];
  const float* we2    = (const float*)d_in[20];
  const float* wr     = (const float*)d_in[21];
  const float* br     = (const float*)d_in[22];
  const float* wbeta  = (const float*)d_in[23];
  const float* lng    = (const float*)d_in[24];
  const float* lnb    = (const float*)d_in[25];
  const int*   ei     = (const int*)d_in[26];
  (void)in_sizes; (void)n_in; (void)ws_size; (void)out_size;

  char* ws = (char*)d_ws;
  size_t off = 0;
  auto alloc = [&](size_t bytes) -> void* {
    void* p = ws + off;
    off += (bytes + 255) & ~(size_t)255;
    return p;
  };

  __bf16* Wqb[4]; __bf16* Wkb[4]; __bf16* Wvb[4]; __bf16* Wrb[4];
  int wElems[4] = {64 * DD, DD * DD, DD * DD, DD * DD};
  for (int l = 0; l < 4; ++l) {
    Wqb[l] = (__bf16*)alloc(wElems[l] * 2);
    Wkb[l] = (__bf16*)alloc(wElems[l] * 2);
    Wvb[l] = (__bf16*)alloc(wElems[l] * 2);
    Wrb[l] = (__bf16*)alloc(wElems[l] * 2);
  }
  __bf16* xb    = (__bf16*)alloc((size_t)NN * DD * 2);
  __bf16* qb    = (__bf16*)alloc((size_t)NN * DD * 2);
  __bf16* kb    = (__bf16*)alloc((size_t)NN * DD * 2);
  __bf16* vbt   = (__bf16*)alloc((size_t)NN * DD * 2);
  float* qf     = (float*)alloc((size_t)NN * DD * 4);
  float* kf     = (float*)alloc((size_t)NN * DD * 4);
  float* vf     = (float*)alloc((size_t)NN * DD * 4);
  float* rf     = (float*)alloc((size_t)NN * DD * 4);
  float* attn   = (float*)alloc((size_t)NN * DD * 4);
  float* hbuf   = (float*)alloc((size_t)NN * DD * 4);
  float* alphaB = (float*)alloc((size_t)EE * HH * 4);
  float* amax   = (float*)alloc((size_t)NN * HH * 4);
  float* denom  = (float*)alloc((size_t)NN * HH * 4);

  auto cvt = [&](const float* s, __bf16* d, int n) {
    mat_f32_to_bf16<<<(n + 255) / 256, 256, 0, stream>>>(s, d, n);
  };
  auto cvtWT = [&](const float* s, __bf16* d, int K) {
    wt_transpose_bf16<<<(K * DD + 255) / 256, 256, 0, stream>>>(s, d, K);
  };

  const float* Wq_src[4] = {wq0, wq, wq + DD * DD, wq + 2 * DD * DD};
  const float* Wk_src[4] = {wk0, wk, wk + DD * DD, wk + 2 * DD * DD};
  const float* Wv_src[4] = {wv0, wv, wv + DD * DD, wv + 2 * DD * DD};
  const float* Wr_src[4] = {wr0, wr, wr + DD * DD, wr + 2 * DD * DD};
  const float* Bq_src[4] = {bq0, bq, bq + DD, bq + 2 * DD};
  const float* Bk_src[4] = {bk0, bk, bk + DD, bk + 2 * DD};
  const float* Bv_src[4] = {bv0, bv, bv + DD, bv + 2 * DD};
  const float* Br_src[4] = {br0, br, br + DD, br + 2 * DD};
  const float* Wb_src[4] = {wbeta0, wbeta, wbeta + 3 * DD, wbeta + 6 * DD};
  const float* Lng_src[4] = {lng0, lng, lng + DD, lng + 2 * DD};
  const float* Lnb_src[4] = {lnb0, lnb, lnb + DD, lnb + 2 * DD};
  const float* We_src[4] = {we0, nullptr, we2, nullptr};
  const int Kdim[4] = {64, DD, DD, DD};
  const bool denseL[4] = {false, true, false, true};

  for (int l = 0; l < 4; ++l) {
    cvtWT(Wq_src[l], Wqb[l], Kdim[l]);
    cvtWT(Wk_src[l], Wkb[l], Kdim[l]);
    cvtWT(Wv_src[l], Wvb[l], Kdim[l]);
    cvtWT(Wr_src[l], Wrb[l], Kdim[l]);
  }

  for (int l = 0; l < 4; ++l) {
    const float* xin = (l == 0) ? x : hbuf;
    cvt(xin, xb, NN * Kdim[l]);

    dim3 ggrid(NN / 128, DD / 64);
    gemm_bias_wmma<<<ggrid, 256, 0, stream>>>(xb, Wqb[l], Bq_src[l], qf, Kdim[l]);
    gemm_bias_wmma<<<ggrid, 256, 0, stream>>>(xb, Wkb[l], Bk_src[l], kf, Kdim[l]);
    gemm_bias_wmma<<<ggrid, 256, 0, stream>>>(xb, Wvb[l], Bv_src[l], vf, Kdim[l]);
    gemm_bias_wmma<<<ggrid, 256, 0, stream>>>(xb, Wrb[l], Br_src[l], rf, Kdim[l]);

    const float* denomPtr = nullptr;
    if (denseL[l]) {
      cvt(qf, qb, NN * DD);
      cvt(kf, kb, NN * DD);
      v_transpose_bf16<<<(NN * DD + 255) / 256, 256, 0, stream>>>(vf, vbt);
      dim3 agrid(GG, HH);
      dense_attn_wmma<<<agrid, 32, 0, stream>>>(qb, kb, vbt, attn);
    } else {
      fill_f32<<<(NN * HH + 255) / 256, 256, 0, stream>>>(amax, -3.0e38f, NN * HH);
      fill_f32<<<(NN * HH + 255) / 256, 256, 0, stream>>>(denom, 0.f, NN * HH);
      fill_f32<<<(NN * DD + 255) / 256, 256, 0, stream>>>(attn, 0.f, NN * DD);
      sparse_alpha<<<(EE * HH + 255) / 256, 256, 0, stream>>>(qf, kf, ei, eattr, We_src[l],
                                                              alphaB, amax);
      sparse_softmax_p<<<(EE * HH + 255) / 256, 256, 0, stream>>>(ei, alphaB, amax, denom);
      sparse_accum<<<EE, 256, 0, stream>>>(ei, eattr, We_src[l], vf, alphaB, attn);
      denomPtr = denom;
    }

    float* dst = (l == 3) ? (float*)d_out : hbuf;
    node_epilogue<<<NN, 256, 0, stream>>>(attn, denomPtr, rf, Wb_src[l],
                                          Lng_src[l], Lnb_src[l], dst);
  }
}
